// QuaternionDense_60584808677734
// MI455X (gfx1250) — matrix operand order repaired and checked by reading, not compiled, not run on hardware
//
#include <hip/hip_runtime.h>

typedef __attribute__((ext_vector_type(16))) __bf16 v16bf;
typedef __attribute__((ext_vector_type(8)))  __bf16 v8bf;
typedef __attribute__((ext_vector_type(8)))  float  v8f;

#define IN_F   1024
#define OUT_F  1024
#define BATCH  8192
#define K_DIM  (4 * IN_F)    // 4096 (reduction dim)
#define N_DIM  (4 * OUT_F)   // 4096 (output features)

#define TM 256               // output rows per block
#define TN 128               // output cols per block
#define KC 64                // K chunk staged in LDS
#define LDSS 72              // padded LDS row stride (elements); 144B keeps 16B alignment

// ---------------------------------------------------------------------------
// Pass 1a: x fp32 -> bf16
// ---------------------------------------------------------------------------
__global__ void cvt_x_bf16(const float* __restrict__ x, __bf16* __restrict__ xb, int n8) {
    int idx = blockIdx.x * blockDim.x + threadIdx.x;
    int stride = gridDim.x * blockDim.x;
    for (int i = idx; i < n8; i += stride) {
        const float4* p = reinterpret_cast<const float4*>(x) + 2 * (size_t)i;
        float4 a = p[0], b = p[1];
        v8bf o;
        o[0] = (__bf16)a.x; o[1] = (__bf16)a.y; o[2] = (__bf16)a.z; o[3] = (__bf16)a.w;
        o[4] = (__bf16)b.x; o[5] = (__bf16)b.y; o[6] = (__bf16)b.z; o[7] = (__bf16)b.w;
        reinterpret_cast<v8bf*>(xb)[i] = o;
    }
}

// ---------------------------------------------------------------------------
// Pass 1b: build bf16 W_block[4096][4096] with Hamilton sign pattern.
//   component index = q_o ^ q_i ; negatives at (0,1)(0,2)(0,3)(1,3)(2,1)(3,2)
// ---------------------------------------------------------------------------
__global__ void build_wblock_bf16(const float* __restrict__ Wr, const float* __restrict__ Wi,
                                  const float* __restrict__ Wj, const float* __restrict__ Wk,
                                  __bf16* __restrict__ wb) {
    const int total8 = (N_DIM * K_DIM) / 8;
    int idx = blockIdx.x * blockDim.x + threadIdx.x;
    int stride = gridDim.x * blockDim.x;
    for (int t = idx; t < total8; t += stride) {
        int e  = t * 8;
        int o  = e >> 12;
        int i  = e & (K_DIM - 1);
        int qo = o >> 10, oo = o & (OUT_F - 1);
        int qi = i >> 10, ii = i & (IN_F - 1);
        int c  = qo ^ qi;
        const float* src = (c == 0) ? Wr : (c == 1) ? Wi : (c == 2) ? Wj : Wk;
        src += (size_t)oo * IN_F + ii;
        float s = ((0x428E >> ((qo << 2) | qi)) & 1) ? -1.0f : 1.0f;
        const float4* p = reinterpret_cast<const float4*>(src);
        float4 a = p[0], b = p[1];
        v8bf ov;
        ov[0] = (__bf16)(s * a.x); ov[1] = (__bf16)(s * a.y);
        ov[2] = (__bf16)(s * a.z); ov[3] = (__bf16)(s * a.w);
        ov[4] = (__bf16)(s * b.x); ov[5] = (__bf16)(s * b.y);
        ov[6] = (__bf16)(s * b.z); ov[7] = (__bf16)(s * b.w);
        reinterpret_cast<v8bf*>(wb)[t] = ov;
    }
}

// ---------------------------------------------------------------------------
// Pass 2: bf16 NT GEMM with V_WMMA_F32_16X16X32_BF16.
//   Block: 256 threads = 8 wave32, 256x128 tile. Wave grid 4(M) x 2(N);
//   each wave owns 64x64 = 4x4 accumulators.
//   Pipeline: [global->regs (next chunk)] overlaps [LDS->wmma (this chunk)],
//   regs -> ds_store into the other LDS buffer, ONE barrier per chunk.
// ---------------------------------------------------------------------------
__launch_bounds__(256)
__global__ void qgemm_bf16_wmma(const __bf16* __restrict__ X, const __bf16* __restrict__ W,
                                const float* __restrict__ bias, float* __restrict__ out) {
    __shared__ __align__(16) __bf16 Xs[2][TM][LDSS];   // 2 x 36 KB
    __shared__ __align__(16) __bf16 Wt[2][TN][LDSS];   // 2 x 18 KB

    const int t    = threadIdx.x;
    const int lane = t & 31;
    const int w    = t >> 5;       // wave id 0..7
    const int wm   = w & 3;        // 4 waves in M (64 rows each)
    const int wn   = w >> 2;       // 2 waves in N (64 cols each)
    const int half = lane >> 4;
    const int l16  = lane & 15;

    const int m_base = blockIdx.y * TM;
    const int n_base = blockIdx.x * TN;

    // staging geometry: 16B per thread per iteration, 8 threads cover one 64-col row
    const int srowX[8] = { (t + 0*256) >> 3, (t + 1*256) >> 3, (t + 2*256) >> 3, (t + 3*256) >> 3,
                           (t + 4*256) >> 3, (t + 5*256) >> 3, (t + 6*256) >> 3, (t + 7*256) >> 3 };
    const int scc = (t & 7) * 8;

    v8f acc[4][4] = {};
    int4 rx[8];   // staged X regs: 256x64 bf16 / 256 thr = 8 x 16B
    int4 rw[4];   // staged W regs: 128x64 bf16 / 256 thr = 4 x 16B

    // ---- prologue: load chunk 0 into regs, store to LDS buffer 0 ----
#pragma unroll
    for (int it = 0; it < 8; ++it)
        rx[it] = *reinterpret_cast<const int4*>(X + (size_t)(m_base + srowX[it]) * K_DIM + scc);
#pragma unroll
    for (int it = 0; it < 4; ++it)
        rw[it] = *reinterpret_cast<const int4*>(W + (size_t)(n_base + srowX[it]) * K_DIM + scc);
#pragma unroll
    for (int it = 0; it < 8; ++it)
        *reinterpret_cast<int4*>(&Xs[0][srowX[it]][scc]) = rx[it];
#pragma unroll
    for (int it = 0; it < 4; ++it)
        *reinterpret_cast<int4*>(&Wt[0][srowX[it]][scc]) = rw[it];
    __syncthreads();

    int p = 0;
    for (int kc = 0; kc < K_DIM; kc += KC) {
        const bool more = (kc + KC) < K_DIM;
        // ---- issue global loads for the NEXT chunk (latency hidden by wmmas) ----
        if (more) {
#pragma unroll
            for (int it = 0; it < 8; ++it)
                rx[it] = *reinterpret_cast<const int4*>(
                    X + (size_t)(m_base + srowX[it]) * K_DIM + kc + KC + scc);
#pragma unroll
            for (int it = 0; it < 4; ++it)
                rw[it] = *reinterpret_cast<const int4*>(
                    W + (size_t)(n_base + srowX[it]) * K_DIM + kc + KC + scc);
        }

        // ---- compute on buffer p: 2 K-steps x 16 wmmas ----
#pragma unroll
        for (int kk = 0; kk < KC; kk += 32) {
            // preload ALL fragments into distinct live values, then burst wmmas
            v16bf afrag[4], bfrag[4];
#pragma unroll
            for (int mt = 0; mt < 4; ++mt) {
                const __bf16* base = &Xs[p][wm * 64 + mt * 16 + l16][kk + 8 * half];
                v8bf lo = *reinterpret_cast<const v8bf*>(base);
                v8bf hi = *reinterpret_cast<const v8bf*>(base + 16);
                afrag[mt] = __builtin_shufflevector(lo, hi,
                    0, 1, 2, 3, 4, 5, 6, 7, 8, 9, 10, 11, 12, 13, 14, 15);
            }
#pragma unroll
            for (int nt = 0; nt < 4; ++nt) {
                const __bf16* base = &Wt[p][wn * 64 + nt * 16 + l16][kk + 8 * half];
                v8bf lo = *reinterpret_cast<const v8bf*>(base);
                v8bf hi = *reinterpret_cast<const v8bf*>(base + 16);
                bfrag[nt] = __builtin_shufflevector(lo, hi,
                    0, 1, 2, 3, 4, 5, 6, 7, 8, 9, 10, 11, 12, 13, 14, 15);
            }
#pragma unroll
            for (int nt = 0; nt < 4; ++nt)
#pragma unroll
                for (int mt = 0; mt < 4; ++mt)
                    acc[mt][nt] = __builtin_amdgcn_wmma_f32_16x16x32_bf16(
                        false, afrag[mt], false, bfrag[nt],
                        (short)0, acc[mt][nt], false, false);
        }

        // ---- write staged regs into the other buffer; one barrier per chunk ----
        if (more) {
#pragma unroll
            for (int it = 0; it < 8; ++it)
                *reinterpret_cast<int4*>(&Xs[p ^ 1][srowX[it]][scc]) = rx[it];
#pragma unroll
            for (int it = 0; it < 4; ++it)
                *reinterpret_cast<int4*>(&Wt[p ^ 1][srowX[it]][scc]) = rw[it];
        }
        __syncthreads();
        p ^= 1;
    }

    // ---- epilogue: D layout — VGPR v = row v (lanes 0-15) / v+8 (lanes 16-31),
    //      col = lane&15. Add bias, store fp32. ----
#pragma unroll
    for (int nt = 0; nt < 4; ++nt) {
        int col  = n_base + wn * 64 + nt * 16 + l16;
        float bv = bias[col];
#pragma unroll
        for (int mt = 0; mt < 4; ++mt) {
            int row0 = m_base + wm * 64 + mt * 16 + 8 * half;
#pragma unroll
            for (int v = 0; v < 8; ++v) {
                out[(size_t)(row0 + v) * N_DIM + col] = acc[mt][nt][v] + bv;
            }
        }
    }
}

// ---------------------------------------------------------------------------
extern "C" void kernel_launch(void* const* d_in, const int* in_sizes, int n_in,
                              void* d_out, int out_size, void* d_ws, size_t ws_size,
                              hipStream_t stream) {
    const float* x    = (const float*)d_in[0];
    const float* Wr   = (const float*)d_in[1];
    const float* Wi   = (const float*)d_in[2];
    const float* Wj   = (const float*)d_in[3];
    const float* Wk   = (const float*)d_in[4];
    const float* bias = (const float*)d_in[5];
    float* out = (float*)d_out;

    // workspace layout: [ x_bf16 : 64MB ][ W_block_bf16 : 32MB ]
    __bf16* xb = (__bf16*)d_ws;
    __bf16* wb = (__bf16*)((char*)d_ws + (size_t)BATCH * K_DIM * sizeof(__bf16));

    const int n8x = (BATCH * K_DIM) / 8;
    cvt_x_bf16<<<2048, 256, 0, stream>>>(x, xb, n8x);
    build_wblock_bf16<<<2048, 256, 0, stream>>>(Wr, Wi, Wj, Wk, wb);

    dim3 grid(N_DIM / TN, BATCH / TM);   // (32, 32)
    qgemm_bf16_wmma<<<grid, 256, 0, stream>>>(xb, wb, bias, out);
}
